// GBNet_70738111365446
// MI455X (gfx1250) — compile-verified, matrix operand order checked by reading, not run on hardware
//
#include <hip/hip_runtime.h>

// ---------------------------------------------------------------------------
// GBNet forward for MI455X (gfx1250, wave32).
// All matmuls run on v_wmma_f32_16x16x32_f16 (f16 A/B, f32 accumulate).
// Large GEMMs use a 64x64-per-wave register tile (4x4 accumulators,
// 16 WMMAs per K-step, 32 FLOP/byte); skinny GEMMs fall back to 16x16.
// BN / softmax / topk / pooling stay in f32 VALU.
// ---------------------------------------------------------------------------

#define BB   16       // batch
#define NP   1024     // points
#define KNN  20       // knn neighbors
#define NEG  0.2f     // LeakyReLU slope
#define BN_EPS 1e-5f

typedef __attribute__((ext_vector_type(16))) _Float16 v16h;
typedef __attribute__((ext_vector_type(8)))  _Float16 v8h;
typedef __attribute__((ext_vector_type(8)))  float    v8f;

static inline int pad16(int x) { return (x + 15) & ~15; }
static inline int pad32(int x) { return (x + 31) & ~31; }

__device__ __forceinline__ v16h frag_load(const _Float16* p)
{
    v8h lo, hi;
    __builtin_memcpy(&lo, p, 16);
    __builtin_memcpy(&hi, p + 16, 16);
    return __builtin_shufflevector(lo, hi, 0,1,2,3,4,5,6,7,8,9,10,11,12,13,14,15);
}

// ---------------------------------------------------------------------------
// 16x16-tile WMMA GEMM (fallback for skinny shapes):
// D[b] (MxN f32, ldd) = A[b] (MxK f16, lda, row-major) x Bt[b] (NxK f16, ldb).
// M,N mult 16; K,lda,ldb mult 32 (zero padded).
// ---------------------------------------------------------------------------
__global__ __launch_bounds__(32)
void gemm16(const _Float16* __restrict__ A, const _Float16* __restrict__ B,
            float* __restrict__ D, int Nt, int Kred,
            int lda, int ldb, int ldd,
            long long sA, long long sB, long long sD)
{
    const long long b = blockIdx.y;
    const int tm = blockIdx.x / Nt, tn = blockIdx.x % Nt;
    const int lane = threadIdx.x & 31;
    const int r = lane & 15, hf = lane >> 4;

    const _Float16* pa = A + b * sA + (long long)(tm * 16 + r) * lda + 8 * hf;
    const _Float16* pb = B + b * sB + (long long)(tn * 16 + r) * ldb + 8 * hf;

    v8f acc = {0.f, 0.f, 0.f, 0.f, 0.f, 0.f, 0.f, 0.f};

    for (int k = 0; k < Kred; k += 32) {
        __builtin_prefetch(pa + k + 64, 0, 3);   // global_prefetch_b8
        __builtin_prefetch(pb + k + 64, 0, 3);
        v16h av = frag_load(pa + k);
        v16h bv = frag_load(pb + k);
        acc = __builtin_amdgcn_wmma_f32_16x16x32_f16(false, av, false, bv,
                                                     (short)0, acc, false, false);
    }

    float* pd = D + b * sD + (long long)(tm * 16 + 8 * hf) * ldd + tn * 16 + r;
#pragma unroll
    for (int v = 0; v < 8; ++v) pd[(long long)v * ldd] = acc[v];
}

// ---------------------------------------------------------------------------
// 64x64-tile WMMA GEMM: one wave computes 4x4 16x16 subtiles.
// Per K-step: 8 fragment loads (16 x b128) feed 16 v_wmma -> 4x operand reuse.
// Requires M%64==0, N%64==0.
// ---------------------------------------------------------------------------
__global__ __launch_bounds__(32)
void gemm64(const _Float16* __restrict__ A, const _Float16* __restrict__ B,
            float* __restrict__ D, int Nt, int Kred,
            int lda, int ldb, int ldd,
            long long sA, long long sB, long long sD)
{
    const long long b = blockIdx.y;
    const int tm = blockIdx.x / Nt, tn = blockIdx.x % Nt;   // Nt = N/64
    const int lane = threadIdx.x & 31;
    const int r = lane & 15, hf = lane >> 4;

    const _Float16* pa = A + b * sA + (long long)(tm * 64 + r) * lda + 8 * hf;
    const _Float16* pb = B + b * sB + (long long)(tn * 64 + r) * ldb + 8 * hf;
    const long long a16 = 16LL * lda, b16 = 16LL * ldb;

    const v8f z = {0.f, 0.f, 0.f, 0.f, 0.f, 0.f, 0.f, 0.f};
    v8f acc[4][4];
#pragma unroll
    for (int i = 0; i < 4; ++i)
#pragma unroll
        for (int j = 0; j < 4; ++j) acc[i][j] = z;

    for (int k = 0; k < Kred; k += 32) {
        __builtin_prefetch(pa + k + 64, 0, 3);
        __builtin_prefetch(pb + k + 64, 0, 3);
        v16h af[4], bf[4];
#pragma unroll
        for (int i = 0; i < 4; ++i) af[i] = frag_load(pa + (long long)i * a16 + k);
#pragma unroll
        for (int j = 0; j < 4; ++j) bf[j] = frag_load(pb + (long long)j * b16 + k);
#pragma unroll
        for (int i = 0; i < 4; ++i)
#pragma unroll
            for (int j = 0; j < 4; ++j)
                acc[i][j] = __builtin_amdgcn_wmma_f32_16x16x32_f16(
                    false, af[i], false, bf[j], (short)0, acc[i][j], false, false);
    }

    float* pd = D + b * sD + (long long)(tm * 64 + 8 * hf) * ldd + tn * 64 + r;
#pragma unroll
    for (int i = 0; i < 4; ++i)
#pragma unroll
        for (int j = 0; j < 4; ++j)
#pragma unroll
            for (int v = 0; v < 8; ++v)
                pd[(long long)(i * 16 + v) * ldd + j * 16] = acc[i][j][v];
}

// ---------------------------------------------------------------------------
// Packing kernels (f32 -> f16, zero-padded)
// ---------------------------------------------------------------------------
__global__ void pack_rm_f16(_Float16* dst, const float* src, int rowsV, int cols,
                            int ld, long long sSrc, long long sDst, int total)
{
    long long b = blockIdx.y;
    int i = blockIdx.x * blockDim.x + threadIdx.x;
    if (i >= total) return;
    int rr = i / ld, c = i % ld;
    float v = (rr < rowsV && c < cols) ? src[b * sSrc + (long long)rr * cols + c] : 0.f;
    dst[b * sDst + i] = (_Float16)v;
}

// dst[b][j*ld + r] = (r < R) ? src[b][r*Cc + j] : 0    (j in [0,Cc), r in [0,ld))
__global__ void pack_tr_f16(_Float16* dst, const float* src, int R, int Cc, int ld,
                            long long sSrc, long long sDst)
{
    long long b = blockIdx.y;
    int i = blockIdx.x * blockDim.x + threadIdx.x;
    if (i >= Cc * ld) return;
    int j = i / ld, r2 = i % ld;
    float v = (r2 < R) ? src[b * sSrc + (long long)r2 * Cc + j] : 0.f;
    dst[b * sDst + i] = (_Float16)v;
}

// dst[b][n*ld + c*K + j] = src[b][(c*N + n)*K + j]   (ck-layout repack of (O,N,K))
__global__ void pack_ck_f16(_Float16* dst, const float* src, int O, int N, int Kk,
                            int ld, long long sSrc, long long sDst)
{
    long long b = blockIdx.y;
    long long i = blockIdx.x * (long long)blockDim.x + threadIdx.x;
    if (i >= (long long)N * ld) return;
    int n = (int)(i / ld), q = (int)(i % ld);
    int c = q / Kk, j = q % Kk;
    float v = (c < O) ? src[b * sSrc + ((long long)c * N + n) * Kk + j] : 0.f;
    dst[b * sDst + i] = (_Float16)v;
}

// ---------------------------------------------------------------------------
// Edge feature packing: writes Bt for the 1x1 conv GEMM (rows = (n,j), ld1)
// and optionally the ck-layout Bt (rows = n, ldck).
// ---------------------------------------------------------------------------
__global__ void edge_pack(_Float16* dT, _Float16* dCk, const float* x, const int* idx,
                          int C, int ld1, int ldck, long long sX)
{
    long long b = blockIdx.y;
    int i = blockIdx.x * blockDim.x + threadIdx.x;
    if (i >= NP * KNN) return;
    int n = i / KNN, j = i % KNN;
    int m = idx[(b * NP + n) * KNN + j];
    const float* xb = x + b * sX;
    _Float16* pT = dT + (b * (long long)NP * KNN + i) * ld1;
    _Float16* pC = dCk ? dCk + (b * (long long)NP + n) * ldck : (_Float16*)0;
    for (int c = 0; c < C; ++c) {
        float xc = xb[(long long)c * NP + n];
        float xm = xb[(long long)c * NP + m];
        _Float16 dv = (_Float16)(xm - xc), cv = (_Float16)xc;
        pT[c] = dv; pT[C + c] = cv;
        if (pC) { pC[c * KNN + j] = dv; pC[(C + c) * KNN + j] = cv; }
    }
    for (int c = 2 * C; c < ld1; ++c) pT[c] = (_Float16)0.f;
    if (pC && j == 0)
        for (int q = 2 * C * KNN; q < ldck; ++q) pC[q] = (_Float16)0.f;
}

// ---------------------------------------------------------------------------
// knn helpers
// ---------------------------------------------------------------------------
__global__ void xx_kernel(float* xx, const float* x, int C, long long sX)
{
    long long b = blockIdx.y;
    int n = blockIdx.x * blockDim.x + threadIdx.x;
    if (n >= NP) return;
    const float* p = x + b * sX;
    float s = 0.f;
    for (int c = 0; c < C; ++c) { float v = p[(long long)c * NP + n]; s += v * v; }
    xx[b * NP + n] = s;
}

__global__ void topk20(int* idx, const float* inner, const float* xx)
{
    long long b = blockIdx.y;
    int n = blockIdx.x * blockDim.x + threadIdx.x;
    if (n >= NP) return;
    const float* row = inner + (b * NP + n) * (long long)NP;
    const float* xxb = xx + b * NP;
    float xn = xxb[n];
    float bv[KNN]; int bi[KNN];
#pragma unroll
    for (int t = 0; t < KNN; ++t) { bv[t] = -3.4e38f; bi[t] = 0; }
    for (int m = 0; m < NP; ++m) {
        float v = 2.f * row[m] - xn - xxb[m];
        if (v > bv[KNN - 1]) {
            bv[KNN - 1] = v; bi[KNN - 1] = m;
#pragma unroll
            for (int t = KNN - 1; t > 0; --t)
                if (bv[t] > bv[t - 1]) {
                    float tv = bv[t]; bv[t] = bv[t - 1]; bv[t - 1] = tv;
                    int   ti = bi[t]; bi[t] = bi[t - 1]; bi[t - 1] = ti;
                }
        }
    }
    int* o = idx + (b * NP + n) * KNN;
#pragma unroll
    for (int t = 0; t < KNN; ++t) o[t] = bi[t];
}

// ---------------------------------------------------------------------------
// geometric_point_descriptor: (B,3,N) -> (B,14,N), knn-3 inline
// ---------------------------------------------------------------------------
__global__ void geo_kernel(float* x0, const float* x)
{
    long long b = blockIdx.y;
    int n = blockIdx.x * blockDim.x + threadIdx.x;
    if (n >= NP) return;
    const float* xb = x + b * 3 * NP;
    float px = xb[n], py = xb[NP + n], pz = xb[2 * NP + n];
    float b0 = -3.4e38f, b1 = -3.4e38f, b2 = -3.4e38f; int i0 = 0, i1 = 0, i2 = 0;
    for (int m = 0; m < NP; ++m) {
        float dx = xb[m] - px, dy = xb[NP + m] - py, dz = xb[2 * NP + m] - pz;
        float v = -(dx * dx + dy * dy + dz * dz);
        if (v > b0)      { b2 = b1; i2 = i1; b1 = b0; i1 = i0; b0 = v; i0 = m; }
        else if (v > b1) { b2 = b1; i2 = i1; b1 = v;  i1 = m; }
        else if (v > b2) { b2 = v;  i2 = m; }
    }
    float e1x = xb[i1] - px, e1y = xb[NP + i1] - py, e1z = xb[2 * NP + i1] - pz;
    float e2x = xb[i2] - px, e2y = xb[NP + i2] - py, e2z = xb[2 * NP + i2] - pz;
    float cx = e1y * e2z - e1z * e2y, cy = e1z * e2x - e1x * e2z, cz = e1x * e2y - e1y * e2x;
    float d1 = sqrtf(e1x * e1x + e1y * e1y + e1z * e1z);
    float d2 = sqrtf(e2x * e2x + e2y * e2y + e2z * e2z);
    float* o = x0 + b * 14 * NP;
    o[0*NP+n]=px;  o[1*NP+n]=py;  o[2*NP+n]=pz;
    o[3*NP+n]=cx;  o[4*NP+n]=cy;  o[5*NP+n]=cz;
    o[6*NP+n]=d1;  o[7*NP+n]=d2;
    o[8*NP+n]=e1x; o[9*NP+n]=e1y; o[10*NP+n]=e1z;
    o[11*NP+n]=e2x; o[12*NP+n]=e2y; o[13*NP+n]=e2z;
}

// ---------------------------------------------------------------------------
// BatchNorm (training mode) + (leaky)ReLU
// element (b,c,s) at  b*strideB + c*strideC + s
// ---------------------------------------------------------------------------
__global__ void bn_stats(float* stats, const float* src, long long S, int Bc,
                         long long strideC, long long strideB)
{
    __shared__ float sm[256], sq[256];
    int c = blockIdx.x, tid = threadIdx.x;
    const float* p = src + (long long)c * strideC;
    long long tot = (long long)Bc * S;
    float s1 = 0.f, s2 = 0.f;
    for (long long i = tid; i < tot; i += blockDim.x) {
        long long bb = i / S, ss = i % S;
        float v = p[bb * strideB + ss];
        s1 += v; s2 += v * v;
    }
    sm[tid] = s1; sq[tid] = s2; __syncthreads();
    for (int o = 128; o > 0; o >>= 1) {
        if (tid < o) { sm[tid] += sm[tid + o]; sq[tid] += sq[tid + o]; }
        __syncthreads();
    }
    if (tid == 0) {
        float m = sm[0] / (float)tot;
        stats[2 * c] = m;
        stats[2 * c + 1] = sq[0] / (float)tot - m * m;
    }
}

__global__ void bn_act(float* dst, const float* src, const float* stats,
                       const float* g, const float* bt, int C, long long S,
                       long long strideC, long long strideB, float slope)
{
    long long b = blockIdx.y;
    long long i = blockIdx.x * (long long)blockDim.x + threadIdx.x;
    if (i >= (long long)C * S) return;
    int c = (int)(i / S); long long s = i % S;
    float v = src[b * strideB + (long long)c * strideC + s];
    float y = (v - stats[2 * c]) * rsqrtf(stats[2 * c + 1] + BN_EPS) * g[c] + bt[c];
    dst[b * strideB + (long long)c * strideC + s] = (y >= 0.f) ? y : slope * y;
}

// softmax(max - sim) over last axis == softmax(-sim); write f16 A-operand
__global__ void softmax_aff(_Float16* aff, const float* sim, int C)
{
    long long b = blockIdx.y;
    int c = blockIdx.x * blockDim.x + threadIdx.x;
    if (c >= C) return;
    const float* row = sim + (b * (long long)C + c) * C;
    float mn = 3.4e38f;
    for (int d = 0; d < C; ++d) mn = fminf(mn, row[d]);
    float s = 0.f;
    for (int d = 0; d < C; ++d) s += __expf(mn - row[d]);
    float inv = 1.f / s;
    _Float16* o = aff + (b * (long long)C + c) * C;
    for (int d = 0; d < C; ++d) o[d] = (_Float16)(__expf(mn - row[d]) * inv);
}

__global__ void axpy_alpha(float* dst, const float* av, const float* x, const float* alpha,
                           long long n, long long sD, long long sA, long long sX)
{
    long long b = blockIdx.y;
    long long i = blockIdx.x * (long long)blockDim.x + threadIdx.x;
    if (i >= n) return;
    dst[b * sD + i] = alpha[0] * av[b * sA + i] + x[b * sX + i];
}

__global__ void sub_k(float* dst, const float* a, const float* c, long long n,
                      long long sD, long long sA, long long sC)
{
    long long b = blockIdx.y;
    long long i = blockIdx.x * (long long)blockDim.x + threadIdx.x;
    if (i >= n) return;
    dst[b * sD + i] = a[b * sA + i] - c[b * sC + i];
}

__global__ void maxk_add(float* xm, const float* xa, const float* xc, int O,
                         long long sIn, long long sOut)
{
    long long b = blockIdx.y;
    long long i = blockIdx.x * (long long)blockDim.x + threadIdx.x;
    if (i >= (long long)O * NP) return;
    long long o = i / NP, n = i % NP;
    long long base = b * sIn + o * (long long)NP * KNN + n * KNN;
    float m = -3.4e38f;
#pragma unroll
    for (int j = 0; j < KNN; ++j) m = fmaxf(m, xa[base + j] + xc[base + j]);
    xm[b * sOut + i] = m;
}

__global__ void concat_copy(float* h, const float* src, int C, int cOff)
{
    long long b = blockIdx.y;
    int i = blockIdx.x * blockDim.x + threadIdx.x;
    if (i >= C * NP) return;
    h[b * 1024LL * NP + (long long)cOff * NP + i] = src[b * (long long)C * NP + i];
}

__global__ void pool_kernel(_Float16* pooled, const float* h5)
{
    long long b = blockIdx.y;
    int c = blockIdx.x * blockDim.x + threadIdx.x;
    if (c >= 1024) return;
    const float* p = h5 + (b * 1024 + c) * (long long)NP;
    float mx = -3.4e38f, s = 0.f;
    for (int n = 0; n < NP; ++n) { float v = p[n]; mx = fmaxf(mx, v); s += v; }
    pooled[b * 2048 + c]        = (_Float16)mx;
    pooled[b * 2048 + 1024 + c] = (_Float16)(s * (1.f / NP));
}

__global__ void out_final(float* out, const float* D, const float* bias)
{
    int i = blockIdx.x * blockDim.x + threadIdx.x;
    if (i >= BB * 40) return;
    int b = i / 40, o = i % 40;
    out[i] = D[o * 16 + b] + bias[o];
}

// ---------------------------------------------------------------------------
// Host-side orchestration
// ---------------------------------------------------------------------------
struct ConvP { const float *w, *g, *b; };
struct CaaP  { const float *wq,*gq,*bq,*wk,*gk,*bk,*wv,*gv,*bv,*alpha; };

struct Scr {
    _Float16 *wA, *affA, *xtB, *edgeT, *bigB2, *vT, *qT, *kkT, *pooled, *l1T, *l2T;
    float *inner, *xx, *stats, *xa, *xc, *xb, *tmp, *xl0, *xm, *q, *kk, *sim, *v, *av;
    float *l1, *l2, *l3;
    int* idx;
};

static void gemm(hipStream_t st, const _Float16* A, const _Float16* B, float* D,
                 int M, int Nc, int Kred, int lda, int ldb, int ldd,
                 long long sA, long long sB, long long sD, int batch)
{
    if ((M & 63) == 0 && (Nc & 63) == 0) {
        dim3 g((M / 64) * (Nc / 64), batch);
        gemm64<<<g, 32, 0, st>>>(A, B, D, Nc / 64, Kred, lda, ldb, ldd, sA, sB, sD);
    } else {
        dim3 g((M / 16) * (Nc / 16), batch);
        gemm16<<<g, 32, 0, st>>>(A, B, D, Nc / 16, Kred, lda, ldb, ldd, sA, sB, sD);
    }
}

static void bn(hipStream_t st, Scr& S, float* data, const float* g, const float* b,
               int C, long long Ssp, long long strideC, long long strideB,
               int batch, float slope)
{
    bn_stats<<<dim3(C), 256, 0, st>>>(S.stats, data, Ssp, batch, strideC, strideB);
    long long tot = (long long)C * Ssp;
    bn_act<<<dim3((int)((tot + 255) / 256), batch), 256, 0, st>>>(
        data, data, S.stats, g, b, C, Ssp, strideC, strideB, slope);
}

static void knn_run(hipStream_t st, Scr& S, const float* x, long long sX, int C)
{
    int Cp = pad32(C);
    long long NPCp = (long long)NP * Cp;
    pack_tr_f16<<<dim3((int)((NPCp + 255) / 256), BB), 256, 0, st>>>(
        S.xtB, x, C, NP, Cp, sX, NPCp);
    xx_kernel<<<dim3((NP + 255) / 256, BB), 256, 0, st>>>(S.xx, x, C, sX);
    gemm(st, S.xtB, S.xtB, S.inner, NP, NP, Cp, Cp, Cp, NP,
         NPCp, NPCp, (long long)NP * NP, BB);
    topk20<<<dim3(NP / 256, BB), 256, 0, st>>>(S.idx, S.inner, S.xx);
}

static void caa_run(hipStream_t st, Scr& S, const float* x, long long sX, int C,
                    const CaaP& P, float* out, long long sOut)
{
    long long cs = (long long)C * NP;
    int csb = (int)((cs + 255) / 256);
    // Bt for q,kk: rows = C, ld = 1024 (plain convert of x)
    pack_rm_f16<<<dim3(csb, BB), 256, 0, st>>>(S.xtB, x, C, NP, NP, sX, cs, (int)cs);
    // q = relu(bn(wq (128x1024) x))
    pack_rm_f16<<<dim3((128 * NP + 255) / 256, 1), 256, 0, st>>>(S.wA, P.wq, 128, NP, NP, 0, 0, 128 * NP);
    gemm(st, S.wA, S.xtB, S.q, 128, C, NP, NP, NP, C, 0, cs, (long long)128 * C, BB);
    bn(st, S, S.q, P.gq, P.bq, 128, C, C, (long long)128 * C, BB, 0.f);
    // kk
    pack_rm_f16<<<dim3((128 * NP + 255) / 256, 1), 256, 0, st>>>(S.wA, P.wk, 128, NP, NP, 0, 0, 128 * NP);
    gemm(st, S.wA, S.xtB, S.kk, 128, C, NP, NP, NP, C, 0, cs, (long long)128 * C, BB);
    bn(st, S, S.kk, P.gk, P.bk, 128, C, C, (long long)128 * C, BB, 0.f);
    // sim = kk^T q  : A = kkT (Cx128), Bt = qT (Cx128)
    pack_tr_f16<<<dim3((C * 128 + 255) / 256, BB), 256, 0, st>>>(S.qT,  S.q,  128, C, 128, (long long)128 * C, (long long)C * 128);
    pack_tr_f16<<<dim3((C * 128 + 255) / 256, BB), 256, 0, st>>>(S.kkT, S.kk, 128, C, 128, (long long)128 * C, (long long)C * 128);
    gemm(st, S.kkT, S.qT, S.sim, C, C, 128, 128, 128, C,
         (long long)C * 128, (long long)C * 128, (long long)C * C, BB);
    softmax_aff<<<dim3((C + 255) / 256, BB), 256, 0, st>>>(S.affA, S.sim, C);
    // v = relu(bn(wv x))
    pack_rm_f16<<<dim3((C * C + 255) / 256, 1), 256, 0, st>>>(S.wA, P.wv, C, C, C, 0, 0, C * C);
    pack_tr_f16<<<dim3(csb, BB), 256, 0, st>>>(S.xtB, x, C, NP, C, sX, cs);
    gemm(st, S.wA, S.xtB, S.v, C, NP, C, C, C, NP, 0, cs, cs, BB);
    bn(st, S, S.v, P.gv, P.bv, C, NP, NP, cs, BB, 0.f);
    // av = aff v ; out = alpha*av + x
    pack_tr_f16<<<dim3(csb, BB), 256, 0, st>>>(S.vT, S.v, C, NP, C, cs, cs);
    gemm(st, S.affA, S.vT, S.av, C, NP, C, C, C, NP, (long long)C * C, cs, cs, BB);
    axpy_alpha<<<dim3(csb, BB), 256, 0, st>>>(out, S.av, x, P.alpha, cs, sOut, cs, sX);
}

static void run_block(hipStream_t st, Scr& S, const float* x, int Cin, int O,
                      const ConvP& c1p, const ConvP& ck1p, const ConvP& c2p,
                      const ConvP& ck3p, const CaaP& caaA, const CaaP& caaB,
                      float* xo, float* xl)
{
    const int C2 = 2 * Cin;
    const int ld1 = pad32(C2);
    const int ldck = pad32(C2 * KNN);
    const int ldO = O * KNN;              // multiple of 32 for all blocks
    const int Mck = pad16(Cin);
    const long long sxin = (long long)Cin * NP;
    const long long NK = (long long)NP * KNN;
    const long long sEdge = NK * ld1;
    const long long sXa = (long long)O * NK;

    // --- knn + edge features (both GEMM layouts) ---
    knn_run(st, S, x, sxin, Cin);
    edge_pack<<<dim3((NP * KNN + 255) / 256, BB), 256, 0, st>>>(
        S.edgeT, S.bigB2, x, S.idx, Cin, ld1, ldck, sxin);

    // --- xa = c1(edge, conv_i) ---
    pack_rm_f16<<<dim3((O * ld1 + 255) / 256, 1), 256, 0, st>>>(S.wA, c1p.w, O, C2, ld1, 0, 0, O * ld1);
    gemm(st, S.wA, S.edgeT, S.xa, O, (int)NK, ld1, ld1, ld1, (int)NK, 0, sEdge, sXa, BB);
    bn(st, S, S.xa, c1p.g, c1p.b, O, NK, NK, sXa, BB, NEG);

    // --- xl0 = ck(edge, conv_i3) ---
    pack_rm_f16<<<dim3((O * ldck + 255) / 256, 1), 256, 0, st>>>(S.wA, ck3p.w, O, C2 * KNN, ldck, 0, 0, O * ldck);
    gemm(st, S.wA, S.bigB2, S.xl0, O, NP, ldck, ldck, ldck, NP,
         0, (long long)NP * ldck, (long long)O * NP, BB);
    bn(st, S, S.xl0, ck3p.g, ck3p.b, O, NP, NP, (long long)O * NP, BB, NEG);

    // --- xb = ck(xa, conv_i1) ---  (bigB2 now free; reuse for xa ck-layout)
    pack_ck_f16<<<dim3((int)(((long long)NP * ldO + 255) / 256), BB), 256, 0, st>>>(
        S.bigB2, S.xa, O, NP, KNN, ldO, sXa, (long long)NP * ldO);
    pack_rm_f16<<<dim3((Mck * ldO + 255) / 256, 1), 256, 0, st>>>(S.wA, ck1p.w, Cin, O * KNN, ldO, 0, 0, Mck * ldO);
    gemm(st, S.wA, S.bigB2, S.xb, Mck, NP, ldO, ldO, ldO, NP,
         0, (long long)NP * ldO, (long long)Mck * NP, BB);
    bn(st, S, S.xb, ck1p.g, ck1p.b, Cin, NP, NP, (long long)Mck * NP, BB, NEG);

    // --- tmp = xb - x_in;  knn2 + second edge set ---
    sub_k<<<dim3((int)((sxin + 255) / 256), BB), 256, 0, st>>>(
        S.tmp, S.xb, x, sxin, sxin, (long long)Mck * NP, sxin);
    knn_run(st, S, S.tmp, sxin, Cin);
    edge_pack<<<dim3((NP * KNN + 255) / 256, BB), 256, 0, st>>>(
        S.edgeT, (_Float16*)0, S.tmp, S.idx, Cin, ld1, 0, sxin);

    // --- xc = c1(edge2, conv_i2) ---
    pack_rm_f16<<<dim3((O * ld1 + 255) / 256, 1), 256, 0, st>>>(S.wA, c2p.w, O, C2, ld1, 0, 0, O * ld1);
    gemm(st, S.wA, S.edgeT, S.xc, O, (int)NK, ld1, ld1, ld1, (int)NK, 0, sEdge, sXa, BB);
    bn(st, S, S.xc, c2p.g, c2p.b, O, NK, NK, sXa, BB, NEG);

    // --- xm = max_k(xa + xc);  two CAA heads ---
    maxk_add<<<dim3((O * NP + 255) / 256, BB), 256, 0, st>>>(S.xm, S.xa, S.xc, O, sXa, (long long)O * NP);
    caa_run(st, S, S.xm,  (long long)O * NP, O, caaA, xo, (long long)O * NP);
    caa_run(st, S, S.xl0, (long long)O * NP, O, caaB, xl, (long long)O * NP);
}

// ---------------------------------------------------------------------------
extern "C" void kernel_launch(void* const* d_in, const int* in_sizes, int n_in,
                              void* d_out, int out_size, void* d_ws, size_t ws_size,
                              hipStream_t stream)
{
    (void)in_sizes; (void)n_in; (void)out_size; (void)ws_size;
    const float* x = (const float*)d_in[0];
    int pi = 1;
    auto nxt = [&]() -> const float* { return (const float*)d_in[pi++]; };
    auto getConv = [&]() { ConvP c; c.w = nxt(); c.g = nxt(); c.b = nxt(); return c; };
    auto getCaa = [&]() {
        CaaP c; c.wq = nxt(); c.gq = nxt(); c.bq = nxt();
        c.wk = nxt(); c.gk = nxt(); c.bk = nxt();
        c.wv = nxt(); c.gv = nxt(); c.bv = nxt(); c.alpha = nxt(); return c;
    };
    ConvP conv1 = getConv(), conv11 = getConv(), conv12 = getConv(), conv13 = getConv();
    CaaP  caa1 = getCaa(), caa11 = getCaa();
    ConvP conv2 = getConv(), conv21 = getConv(), conv22 = getConv(), conv23 = getConv();
    CaaP  caa2 = getCaa(), caa21 = getCaa();
    ConvP conv3 = getConv(), conv31 = getConv(), conv32 = getConv(), conv33 = getConv();
    CaaP  caa3 = getCaa(), caa31 = getCaa();
    ConvP conv4 = getConv(), conv41 = getConv(), conv42 = getConv(), conv43 = getConv();
    CaaP  caa4 = getCaa(), caa41 = getCaa();
    ConvP conv5 = getConv();
    CaaP  caa5 = getCaa();
    ConvP lin1 = getConv();
    const float* l2w = nxt(); const float* l2bias = nxt();   // bias cancels in BN
    const float* l2g = nxt(); const float* l2b = nxt();
    const float* l3w = nxt(); const float* l3bias = nxt();
    (void)l2bias;

    // ---- workspace carve (bump allocator, 256B aligned) ----
    char* base = (char*)d_ws; size_t off = 0;
    auto take = [&](size_t bytes) -> char* {
        off = (off + 255) & ~(size_t)255;
        char* p = base + off; off += bytes; return p;
    };
    const size_t MB = 1024 * 1024;
    float* x0  = (float*)take((size_t)BB * 14  * NP * 4);
    float* x1  = (float*)take((size_t)BB * 64  * NP * 4);
    float* x1l = (float*)take((size_t)BB * 64  * NP * 4);
    float* x2  = (float*)take((size_t)BB * 64  * NP * 4);
    float* x2l = (float*)take((size_t)BB * 64  * NP * 4);
    float* x3  = (float*)take((size_t)BB * 128 * NP * 4);
    float* x3l = (float*)take((size_t)BB * 128 * NP * 4);
    float* x4  = (float*)take((size_t)BB * 256 * NP * 4);
    float* x4l = (float*)take((size_t)BB * 256 * NP * 4);
    float* h   = (float*)take(64 * MB);
    float* h2  = (float*)take(64 * MB);
    float* h5  = (float*)take(64 * MB);

    Scr S;
    S.idx   = (int*)take((size_t)BB * NP * KNN * 4);
    S.xx    = (float*)take((size_t)BB * NP * 4);
    S.stats = (float*)take(2 * 1024 * 4);
    S.inner = (float*)take(64 * MB);
    S.wA    = (_Float16*)take(4 * MB);
    S.affA  = (_Float16*)take(32 * MB);
    S.xtB   = (_Float16*)take(32 * MB);
    S.edgeT = (_Float16*)take(160 * MB);
    S.bigB2 = (_Float16*)take(160 * MB);
    S.qT    = (_Float16*)take(4 * MB);
    S.kkT   = (_Float16*)take(4 * MB);
    S.vT    = (_Float16*)take(32 * MB);
    S.q     = (float*)take(8 * MB);
    S.kk    = (float*)take(8 * MB);
    S.sim   = (float*)take(64 * MB);
    S.v     = (float*)take(64 * MB);
    S.av    = (float*)take(64 * MB);
    S.xa    = (float*)take(320 * MB);
    S.xc    = (float*)take(320 * MB);
    S.xb    = (float*)take(8 * MB);
    S.tmp   = (float*)take(8 * MB);
    S.xl0   = (float*)take(16 * MB);
    S.xm    = (float*)take(16 * MB);
    S.pooled= (_Float16*)take((size_t)BB * 2048 * 2);
    S.l1    = (float*)take(512 * 16 * 4);
    S.l2    = (float*)take(256 * 16 * 4);
    S.l3    = (float*)take(48 * 16 * 4);
    S.l1T   = (_Float16*)take(16 * 512 * 2);
    S.l2T   = (_Float16*)take(16 * 256 * 2);

    // ---- forward pass ----
    geo_kernel<<<dim3((NP + 255) / 256, BB), 256, 0, stream>>>(x0, x);

    run_block(stream, S, x0, 14,  64,  conv1, conv11, conv12, conv13, caa1, caa11, x1, x1l);
    run_block(stream, S, x1, 64,  64,  conv2, conv21, conv22, conv23, caa2, caa21, x2, x2l);
    run_block(stream, S, x2, 64,  128, conv3, conv31, conv32, conv33, caa3, caa31, x3, x3l);
    run_block(stream, S, x3, 128, 256, conv4, conv41, conv42, conv43, caa4, caa41, x4, x4l);

    // concat -> h (B,1024,N)
    concat_copy<<<dim3((64 * NP + 255) / 256, BB), 256, 0, stream>>>(h, x1,  64,  0);
    concat_copy<<<dim3((64 * NP + 255) / 256, BB), 256, 0, stream>>>(h, x1l, 64,  64);
    concat_copy<<<dim3((64 * NP + 255) / 256, BB), 256, 0, stream>>>(h, x2,  64,  128);
    concat_copy<<<dim3((64 * NP + 255) / 256, BB), 256, 0, stream>>>(h, x2l, 64,  192);
    concat_copy<<<dim3((128 * NP + 255) / 256, BB), 256, 0, stream>>>(h, x3,  128, 256);
    concat_copy<<<dim3((128 * NP + 255) / 256, BB), 256, 0, stream>>>(h, x3l, 128, 384);
    concat_copy<<<dim3((256 * NP + 255) / 256, BB), 256, 0, stream>>>(h, x4,  256, 512);
    concat_copy<<<dim3((256 * NP + 255) / 256, BB), 256, 0, stream>>>(h, x4l, 256, 768);

    caa_run(stream, S, h, 1024LL * NP, 1024, caa5, h2, 1024LL * NP);

    // conv5: (1024x1024) x (1024xN) + BN + lrelu
    pack_tr_f16<<<dim3((1024 * 1024 + 255) / 256, BB), 256, 0, stream>>>(
        S.xtB, h2, 1024, NP, 1024, 1024LL * NP, 1024LL * NP);
    pack_rm_f16<<<dim3((1024 * 1024 + 255) / 256, 1), 256, 0, stream>>>(
        S.wA, conv5.w, 1024, 1024, 1024, 0, 0, 1024 * 1024);
    gemm(stream, S.wA, S.xtB, h5, 1024, NP, 1024, 1024, 1024, NP,
         0, 1024LL * NP, 1024LL * NP, BB);
    bn(stream, S, h5, conv5.g, conv5.b, 1024, NP, NP, 1024LL * NP, BB, NEG);

    // pooling: [max, mean] -> (16 x 2048) f16 Bt for lin1
    pool_kernel<<<dim3((1024 + 255) / 256, BB), 256, 0, stream>>>(S.pooled, h5);

    // lin1: (512x2048) -> BN -> lrelu
    pack_rm_f16<<<dim3((512 * 2048 + 255) / 256, 1), 256, 0, stream>>>(S.wA, lin1.w, 512, 2048, 2048, 0, 0, 512 * 2048);
    gemm(stream, S.wA, S.pooled, S.l1, 512, 16, 2048, 2048, 2048, 16, 0, 0, 0, 1);
    bn(stream, S, S.l1, lin1.g, lin1.b, 512, 16, 16, 0, 1, NEG);
    // lin2: (256x512); per-channel bias cancels in training-mode BN
    pack_tr_f16<<<dim3((16 * 512 + 255) / 256, 1), 256, 0, stream>>>(S.l1T, S.l1, 512, 16, 512, 0, 0);
    pack_rm_f16<<<dim3((256 * 512 + 255) / 256, 1), 256, 0, stream>>>(S.wA, l2w, 256, 512, 512, 0, 0, 256 * 512);
    gemm(stream, S.wA, S.l1T, S.l2, 256, 16, 512, 512, 512, 16, 0, 0, 0, 1);
    bn(stream, S, S.l2, l2g, l2b, 256, 16, 16, 0, 1, NEG);
    // lin3: (40x256) padded to 48 rows, + bias
    pack_tr_f16<<<dim3((16 * 256 + 255) / 256, 1), 256, 0, stream>>>(S.l2T, S.l2, 256, 16, 256, 0, 0);
    pack_rm_f16<<<dim3((48 * 256 + 255) / 256, 1), 256, 0, stream>>>(S.wA, l3w, 40, 256, 256, 0, 0, 48 * 256);
    gemm(stream, S.wA, S.l2T, S.l3, 48, 16, 256, 256, 256, 16, 0, 0, 0, 1);
    out_final<<<dim3(3), 256, 0, stream>>>((float*)d_out, S.l3, l3bias);
}
